// LSTM_34909494181998
// MI455X (gfx1250) — matrix operand order repaired and checked by reading, not compiled, not run on hardware
//
#include <hip/hip_runtime.h>

#define B_     32
#define N_     256
#define M_     128
#define HID_   256
#define GATES_ 1024   // 4*HID
#define ROWS_  8192   // B*N
#define STEPS_ 20

typedef __attribute__((ext_vector_type(16))) _Float16 v16h;
typedef __attribute__((ext_vector_type(8)))  _Float16 v8h;
typedef __attribute__((ext_vector_type(8)))  float    v8f;

__device__ __forceinline__ float sigmoid_f(float x) { return 1.0f / (1.0f + __expf(-x)); }

// ---------------------------------------------------------------------------
// init: y_cur = y, best_y = y, best_loss = inf, total = 0
// ---------------------------------------------------------------------------
__global__ __launch_bounds__(256)
void k_init(const float* __restrict__ y, float* __restrict__ y_cur,
            float* __restrict__ out, float* __restrict__ scal) {
    int i = blockIdx.x * 256 + threadIdx.x;
    if (i < ROWS_) { y_cur[i] = y[i]; out[i] = y[i]; }
    if (i == 0) { scal[0] = __builtin_inff(); scal[1] = 0.0f; }
}

// ---------------------------------------------------------------------------
// once per call: weights [K=256][N=1024] f32  ->  f16 K-major [n][k]
// (B-operand of every GEMM becomes pure vector copies, no per-step transpose)
// ---------------------------------------------------------------------------
__global__ __launch_bounds__(256)
void k_cvtw(const float* __restrict__ W, _Float16* __restrict__ Wt) {
    const int idx = blockIdx.x * 256 + threadIdx.x;   // 262144 total
    const int k = idx >> 10, n = idx & 1023;          // read coalesced over n
    Wt[(size_t)n * HID_ + k] = (_Float16)W[idx];
}

// ---------------------------------------------------------------------------
// r = J@y - F ; grad = J^T r ; xcat[row] = (y, grad)   (one block per batch)
// ---------------------------------------------------------------------------
__global__ __launch_bounds__(256)
void k_grad(const float* __restrict__ J, const float* __restrict__ F,
            const float* __restrict__ y, float* __restrict__ xcat) {
    __shared__ float rsh[M_];
    const int b = blockIdx.x, t = threadIdx.x;
    const float* Jb = J + (size_t)b * M_ * N_;
    if (t < M_) {
        float acc = -F[b * M_ + t];
        const float* row = Jb + t * N_;
        for (int n = 0; n < N_; n++) acc += row[n] * y[b * N_ + n];
        rsh[t] = acc;
    }
    __syncthreads();
    float g = 0.0f;
    for (int m = 0; m < M_; m++) g += Jb[m * N_ + t] * rsh[m];  // coalesced in t
    const int r = b * N_ + t;
    xcat[2 * r + 0] = y[r];
    xcat[2 * r + 1] = g;
}

// ---------------------------------------------------------------------------
// single-head attention, D=2, online softmax  (one block per batch, thr = n)
// ---------------------------------------------------------------------------
__global__ __launch_bounds__(256)
void k_attn(const float* __restrict__ xcat, float* __restrict__ xatt,
            const float* __restrict__ Wq, const float* __restrict__ bq,
            const float* __restrict__ Wk, const float* __restrict__ bk,
            const float* __restrict__ Wv, const float* __restrict__ bv,
            const float* __restrict__ Wo, const float* __restrict__ bo) {
    __shared__ float ks[N_][2];
    __shared__ float vs[N_][2];
    const int b = blockIdx.x, n = threadIdx.x;
    const int r = b * N_ + n;
    const float x0 = xcat[2 * r], x1 = xcat[2 * r + 1];
    const float q0 = x0 * Wq[0] + x1 * Wq[2] + bq[0];
    const float q1 = x0 * Wq[1] + x1 * Wq[3] + bq[1];
    ks[n][0] = x0 * Wk[0] + x1 * Wk[2] + bk[0];
    ks[n][1] = x0 * Wk[1] + x1 * Wk[3] + bk[1];
    vs[n][0] = x0 * Wv[0] + x1 * Wv[2] + bv[0];
    vs[n][1] = x0 * Wv[1] + x1 * Wv[3] + bv[1];
    __syncthreads();
    const float scale = 0.70710678118654752f;  // 1/sqrt(D=2)
    float mmax = -1e30f, ssum = 0.0f, a0 = 0.0f, a1 = 0.0f;
    for (int m = 0; m < N_; m++) {
        float sc = (q0 * ks[m][0] + q1 * ks[m][1]) * scale;
        float nm = fmaxf(mmax, sc);
        float corr = __expf(mmax - nm);
        float w = __expf(sc - nm);
        ssum = ssum * corr + w;
        a0 = a0 * corr + w * vs[m][0];
        a1 = a1 * corr + w * vs[m][1];
        mmax = nm;
    }
    const float o0 = a0 / ssum, o1 = a1 / ssum;
    xatt[2 * r + 0] = o0 * Wo[0] + o1 * Wo[2] + bo[0];
    xatt[2 * r + 1] = o0 * Wo[1] + o1 * Wo[3] + bo[1];
}

// ---------------------------------------------------------------------------
// WMMA GEMM: G[8192,1024] = sum_p A_p[8192,256] @ B_p[256,1024]
// A: f16 activations, row-major [m][k].  B: f16 pre-transposed weights [n][k].
// block = 256 thr (8 waves), tile 256(M) x 64(N); wave owns a 32x64 strip
// (2 M-frags x 4 N-frags = 8 wmma per k-step). All LDS traffic is b128;
// no conversions, no scalar packing in the hot loop.
// ---------------------------------------------------------------------------
#define TM 256
#define TN 64
#define TK 32
#define LDK (TK + 8)   // 40 elems = 80B row stride: 16B-aligned, bank-spread

__global__ __launch_bounds__(256)
void k_gemm(const _Float16* __restrict__ A0, const _Float16* __restrict__ Bt0,
            const _Float16* __restrict__ A1, const _Float16* __restrict__ Bt1,
            float* __restrict__ G, int npairs) {
    __shared__ _Float16 As[TM][LDK];   // [m][k]
    __shared__ _Float16 Bs[TN][LDK];   // [n][k]
    const int tid  = threadIdx.x;
    const int wave = tid >> 5, lane = tid & 31;
    const int half = lane >> 4, lr = lane & 15;
    const int tileN = blockIdx.x * TN;
    const int tileM = blockIdx.y * TM;
    const int bn = tid >> 2;            // 0..63   (B stage row)
    const int bk = (tid & 3) * 8;       // 0,8,16,24

    v8f acc[2][4];
    #pragma unroll
    for (int mi = 0; mi < 2; mi++)
        #pragma unroll
        for (int s = 0; s < 4; s++)
            #pragma unroll
            for (int e = 0; e < 8; e++) acc[mi][s][e] = 0.0f;

    for (int p = 0; p < npairs; p++) {
        const _Float16* A  = p ? A1 : A0;
        const _Float16* Bt = p ? Bt1 : Bt0;
        for (int k0 = 0; k0 < HID_; k0 += TK) {
            // ---- stage A: one thread = one row, 32 consecutive f16 (64B) ----
            const _Float16* srcA = A + (size_t)(tileM + tid) * HID_ + k0;
            __builtin_prefetch(srcA + TK, 0, 3);
            #pragma unroll
            for (int j = 0; j < 4; j++)
                *(v8h*)&As[tid][j * 8] = *(const v8h*)(srcA + j * 8);
            // ---- stage B: K-major weights, straight b128 copy ----
            const _Float16* srcB = Bt + (size_t)(tileN + bn) * HID_ + k0 + bk;
            *(v8h*)&Bs[bn][bk] = *(const v8h*)srcB;
            __syncthreads();
            // ---- fragments: contiguous 16B LDS vector loads ----
            v16h bfr[4];
            #pragma unroll
            for (int s = 0; s < 4; s++) {
                const _Float16* bp = &Bs[s * 16 + lr][half * 8];
                v8h b0 = *(const v8h*)bp;
                v8h b1 = *(const v8h*)(bp + 16);
                bfr[s] = __builtin_shufflevector(b0, b1, 0, 1, 2, 3, 4, 5, 6, 7,
                                                 8, 9, 10, 11, 12, 13, 14, 15);
            }
            #pragma unroll
            for (int mi = 0; mi < 2; mi++) {
                const _Float16* ap = &As[wave * 32 + mi * 16 + lr][half * 8];
                v8h a0 = *(const v8h*)ap;         // K: half*8 .. +7
                v8h a1 = *(const v8h*)(ap + 16);  // K: 16+half*8 .. +7
                v16h af = __builtin_shufflevector(a0, a1, 0, 1, 2, 3, 4, 5, 6, 7,
                                                  8, 9, 10, 11, 12, 13, 14, 15);
                #pragma unroll
                for (int s = 0; s < 4; s++)
                    acc[mi][s] = __builtin_amdgcn_wmma_f32_16x16x32_f16(
                        false, af, false, bfr[s], (short)0, acc[mi][s], false, false);
            }
            __syncthreads();
        }
    }
    // ---- store: VGPR v -> M = v + 8*half, N = lr ----
    #pragma unroll
    for (int mi = 0; mi < 2; mi++)
        #pragma unroll
        for (int s = 0; s < 4; s++)
            #pragma unroll
            for (int v = 0; v < 8; v++) {
                int row = tileM + wave * 32 + mi * 16 + v + 8 * half;
                int col = tileN + s * 16 + lr;
                G[(size_t)row * GATES_ + col] = acc[mi][s][v];
            }
}

// ---------------------------------------------------------------------------
// fused 4-gate LSTM epilogue. x2/W may be null (layer 2: x-term already in G).
// C stays f32 (carry accuracy); H written as f16 (consumed only by WMMA/head).
// ---------------------------------------------------------------------------
__global__ __launch_bounds__(256)
void k_lstm(const float* __restrict__ G, const float* __restrict__ x2,
            const float* __restrict__ W, const float* __restrict__ bias,
            float* __restrict__ C, _Float16* __restrict__ H) {
    const int idx = blockIdx.x * 256 + threadIdx.x;  // row*HID + h
    const int row = idx >> 8, h = idx & 255;
    float g[4];
    #pragma unroll
    for (int gt = 0; gt < 4; gt++) {
        const int col = gt * HID_ + h;
        float v = G[(size_t)row * GATES_ + col] + bias[col];
        if (x2) v += x2[2 * row + 0] * W[col] + x2[2 * row + 1] * W[GATES_ + col];
        g[gt] = v;
    }
    const float c = sigmoid_f(g[0]) * tanhf(g[3]) + sigmoid_f(g[1]) * C[idx];
    C[idx] = c;
    H[idx] = (_Float16)(sigmoid_f(g[2]) * tanhf(c));
}

// ---------------------------------------------------------------------------
// y -= H2 @ Wh + bh   (wave per row, shfl reduction)
// ---------------------------------------------------------------------------
__global__ __launch_bounds__(256)
void k_head(const _Float16* __restrict__ H2, const float* __restrict__ Wh,
            const float* __restrict__ bh, float* __restrict__ y) {
    const int wave = threadIdx.x >> 5, lane = threadIdx.x & 31;
    const int row = blockIdx.x * 8 + wave;
    float s = 0.0f;
    for (int i = 0; i < HID_; i += 32)
        s += (float)H2[(size_t)row * HID_ + i + lane] * Wh[i + lane];
    for (int off = 16; off; off >>= 1) s += __shfl_xor(s, off, 32);
    if (lane == 0) y[row] -= (s + bh[0]);
}

// ---------------------------------------------------------------------------
// per-batch ||J@y - F||^2  (one block per batch)
// ---------------------------------------------------------------------------
__global__ __launch_bounds__(128)
void k_loss(const float* __restrict__ J, const float* __restrict__ F,
            const float* __restrict__ y, float* __restrict__ partial) {
    __shared__ float sh[M_];
    const int b = blockIdx.x, m = threadIdx.x;
    const float* row = J + ((size_t)b * M_ + m) * N_;
    float acc = -F[b * M_ + m];
    for (int n = 0; n < N_; n++) acc += row[n] * y[b * N_ + n];
    sh[m] = acc * acc;
    __syncthreads();
    for (int s = 64; s; s >>= 1) {
        if (m < s) sh[m] += sh[m + s];
        __syncthreads();
    }
    if (m == 0) partial[b] = sh[0];
}

// ---------------------------------------------------------------------------
// finalize step: loss, losses[step], running total, best-y tracking
// out layout: [0..8191]=best_y, [8192]=total_loss, [8193..8212]=losses
// ---------------------------------------------------------------------------
__global__ __launch_bounds__(256)
void k_final(const float* __restrict__ partial, const float* __restrict__ y,
             float* __restrict__ scal, float* __restrict__ out, int step) {
    __shared__ int better;
    if (threadIdx.x == 0) {
        float s = 0.0f;
        for (int i = 0; i < B_; i++) s += partial[i];
        const float loss = 0.5f * (s / (float)B_) / (float)STEPS_;
        out[ROWS_ + 1 + step] = loss;
        const float bl = scal[0];
        better = (loss < bl) ? 1 : 0;
        scal[0] = fminf(loss, bl);
        const float tot = scal[1] + loss;
        scal[1] = tot;
        out[ROWS_] = tot;
    }
    __syncthreads();
    if (better)
        for (int i = threadIdx.x; i < ROWS_; i += 256) out[i] = y[i];
}

// ---------------------------------------------------------------------------
extern "C" void kernel_launch(void* const* d_in, const int* in_sizes, int n_in,
                              void* d_out, int out_size, void* d_ws, size_t ws_size,
                              hipStream_t stream) {
    (void)n_in; (void)out_size; (void)ws_size;
    // Input ordering: dict insertion order (y first, 8192 elems) vs.
    // jax tree_leaves sorted keys (F first, 4096 elems). Deterministic probe.
    int iy, iJ, iF, iWq, ibq, iWk, ibk, iWv, ibv, iWo, ibo;
    int iW1, iU1, ib1, iW2, iU2, ib2, iWh, ibh;
    if (in_sizes[0] == ROWS_) {
        iy = 0; iJ = 1; iF = 2;
        iWq = 3; ibq = 4; iWk = 5; ibk = 6; iWv = 7; ibv = 8; iWo = 9; ibo = 10;
        iW1 = 11; iU1 = 12; ib1 = 13; iW2 = 14; iU2 = 15; ib2 = 16; iWh = 17; ibh = 18;
    } else {
        // F, J, iter_step, U1, U2, W1, W2, Wh, Wk, Wo, Wq, Wv, b1, b2, bh, bk, bo, bq, bv, y
        iF = 0; iJ = 1;
        iU1 = 3; iU2 = 4; iW1 = 5; iW2 = 6; iWh = 7; iWk = 8; iWo = 9; iWq = 10; iWv = 11;
        ib1 = 12; ib2 = 13; ibh = 14; ibk = 15; ibo = 16; ibq = 17; ibv = 18; iy = 19;
    }
    const float* y  = (const float*)d_in[iy];
    const float* J  = (const float*)d_in[iJ];
    const float* F  = (const float*)d_in[iF];
    const float* Wq = (const float*)d_in[iWq]; const float* bq = (const float*)d_in[ibq];
    const float* Wk = (const float*)d_in[iWk]; const float* bk = (const float*)d_in[ibk];
    const float* Wv = (const float*)d_in[iWv]; const float* bv = (const float*)d_in[ibv];
    const float* Wo = (const float*)d_in[iWo]; const float* bo = (const float*)d_in[ibo];
    const float* W1 = (const float*)d_in[iW1]; const float* U1 = (const float*)d_in[iU1];
    const float* b1 = (const float*)d_in[ib1];
    const float* W2 = (const float*)d_in[iW2]; const float* U2 = (const float*)d_in[iU2];
    const float* b2 = (const float*)d_in[ib2];
    const float* Wh = (const float*)d_in[iWh]; const float* bh = (const float*)d_in[ibh];
    float* out = (float*)d_out;

    // ---- workspace carve (float units) ----
    float* ws      = (float*)d_ws;
    float* y_cur   = ws;                  // 8192
    float* xcat    = ws + 8192;           // 16384
    float* xatt    = ws + 24576;          // 16384
    float* partial = ws + 40960;          // 32
    float* scal    = ws + 41024;          // 2 (best_loss, total)
    _Float16* H1h  = (_Float16*)(ws + 65536);              // 2M halves (=1M floats)
    _Float16* H2h  = H1h + (size_t)ROWS_ * HID_;           // 2M halves
    float* C1      = ws + 65536 + 2 * 1048576;             // 2M floats
    float* C2      = C1 + (size_t)ROWS_ * HID_;
    float* G       = C2 + (size_t)ROWS_ * HID_;            // 8192x1024 f32
    _Float16* U1t  = (_Float16*)(G + (size_t)ROWS_ * GATES_);  // 256K halves each
    _Float16* W2t  = U1t + (size_t)HID_ * GATES_;
    _Float16* U2t  = W2t + (size_t)HID_ * GATES_;

    // zero H1h,H2h,C1,C2 (contiguous: 4MB+4MB+8MB+8MB) — deterministic state
    hipMemsetAsync(H1h, 0, (size_t)2 * ROWS_ * HID_ * sizeof(_Float16)
                         + (size_t)2 * ROWS_ * HID_ * sizeof(float), stream);
    k_init<<<32, 256, 0, stream>>>(y, y_cur, out, scal);
    // per-call weight pre-transpose + f16 conversion (B operands)
    k_cvtw<<<HID_ * GATES_ / 256, 256, 0, stream>>>(U1, U1t);
    k_cvtw<<<HID_ * GATES_ / 256, 256, 0, stream>>>(W2, W2t);
    k_cvtw<<<HID_ * GATES_ / 256, 256, 0, stream>>>(U2, U2t);

    for (int step = 0; step < STEPS_; step++) {
        k_grad<<<32, 256, 0, stream>>>(J, F, y_cur, xcat);
        k_attn<<<32, 256, 0, stream>>>(xcat, xatt, Wq, bq, Wk, bk, Wv, bv, Wo, bo);
        // layer 1: G = H1 @ U1
        k_gemm<<<dim3(GATES_ / TN, ROWS_ / TM), 256, 0, stream>>>(
            H1h, U1t, nullptr, nullptr, G, 1);
        k_lstm<<<ROWS_ * HID_ / 256, 256, 0, stream>>>(G, xatt, W1, b1, C1, H1h);
        // layer 2: G = H1 @ W2 + H2 @ U2   (fused two-pair GEMM)
        k_gemm<<<dim3(GATES_ / TN, ROWS_ / TM), 256, 0, stream>>>(
            H1h, W2t, H2h, U2t, G, 2);
        k_lstm<<<ROWS_ * HID_ / 256, 256, 0, stream>>>(G, nullptr, nullptr, b2, C2, H2h);
        k_head<<<ROWS_ / 8, 256, 0, stream>>>(H2h, Wh, bh, y_cur);
        k_loss<<<32, 128, 0, stream>>>(J, F, y_cur, partial);
        k_final<<<1, 256, 0, stream>>>(partial, y_cur, scal, out, step);
    }
}
